// MultiHeadAttention_53171695124989
// MI455X (gfx1250) — matrix operand order repaired and checked by reading, not compiled, not run on hardware
//
#include <hip/hip_runtime.h>
#include <hip/hip_bf16.h>

#define DMODEL 1024
#define BATCH  2
#define SEQ    2048
#define NHEAD  16
#define DEPTH  64

typedef __attribute__((ext_vector_type(16))) __bf16 v16bf;
typedef __attribute__((ext_vector_type(8)))  float  v8f;

#if defined(__HIP_DEVICE_COMPILE__) && \
    __has_builtin(__builtin_amdgcn_global_load_async_to_lds_b128)
#define USE_ASYNC_LDS 1
#else
#define USE_ASYNC_LDS 0
#endif

union FragCast { v16bf v; uint4 u[2]; };

// A-matrix fragment (16x32, M x K), row-major source with contraction contiguous.
// Lane L: m = L&15, h = L>>4. Slots 0-7 -> K in [h*8, h*8+8), slots 8-15 -> [16+h*8, 16+h*8+8).
__device__ inline v16bf load_a_frag(const __bf16* base, int stride, int k0) {
  const int lane = threadIdx.x & 31;
  const int m = lane & 15, h = lane >> 4;
  const __bf16* r = base + (size_t)m * stride + k0;
  FragCast f;
  f.u[0] = *(const uint4*)(r + h * 8);
  f.u[1] = *(const uint4*)(r + 16 + h * 8);
  return f.v;
}

// B-matrix fragment (32x16, K x N), gathered from a row-major source where
// source row = N index and columns = contraction K. Lane L: n = L&15, h = L>>4,
// slots 0-7 -> K in [h*8, h*8+8), slots 8-15 -> [16+h*8, 16+h*8+8).
__device__ inline v16bf load_b_frag(const __bf16* base, int stride, int nrow, int k0) {
  const int lane = threadIdx.x & 31;
  const int n = lane & 15, h = lane >> 4;
  const __bf16* r = base + (size_t)(nrow + n) * stride + k0;
  FragCast f;
  f.u[0] = *(const uint4*)(r + h * 8);
  f.u[1] = *(const uint4*)(r + 16 + h * 8);
  return f.v;
}

__device__ inline v8f wmma_bf16(v16bf a, v16bf b, v8f c) {
  return __builtin_amdgcn_wmma_f32_16x16x32_bf16(false, a, false, b, (short)0, c,
                                                 false, false);
}

// ---------------------------------------------------------------------------
// Generic GEMM: C[4096 x 1024] = A[4096 x 1024] @ W[1024 x 1024] + bias
// mode 0: write bf16 head-split [B,H,S,D]            (q, k projections)
// mode 1: write bf16 head-split transposed [B,H,D,S] (v projection)
// mode 2: write fp32 row-major [4096 x 1024]         (final output, NT stores)
// ---------------------------------------------------------------------------
template <bool ABF16>
__global__ __launch_bounds__(256) void gemm_bias_kernel(
    const void* __restrict__ Ap, const float* __restrict__ W,
    const float* __restrict__ bias, void* __restrict__ Out, int mode) {
  __shared__ alignas(16) __bf16 Abuf[128][32];
  __shared__ alignas(16) __bf16 Wt[64][32];

  const int tid = threadIdx.x;
  const int w = tid >> 5, lane = tid & 31;
  const int rowBase = blockIdx.x * 128;
  const int colBase = blockIdx.y * 64;
  const int Kdim = DMODEL, Ndim = DMODEL;

  v8f acc[4] = {};

  const int ldRow = tid >> 1;           // 0..127
  const int ldK   = (tid & 1) * 16;     // 0 or 16
  const int wn    = tid & 63;           // W^T staging: column of W
  const int wk    = (tid >> 6) * 8;     // 0,8,16,24

  for (int k0 = 0; k0 < Kdim; k0 += 32) {
    // Stage A tile (bf16 source -> async DMA to LDS; fp32 source -> convert)
    if constexpr (ABF16) {
      const __bf16* src =
          (const __bf16*)Ap + (size_t)(rowBase + ldRow) * Kdim + k0 + ldK;
#if USE_ASYNC_LDS
      // Builtin expects native vector-of-4-int pointers: AS(1) global src,
      // AS(3) LDS dst. AS(1) shares flat 64-bit addresses; AS(3) is the low
      // 32 bits of the flat LDS address (ISA 10.2).
      typedef int nint4 __attribute__((vector_size(16)));
      typedef __attribute__((address_space(1))) nint4 gnint4;
      typedef __attribute__((address_space(3))) nint4 lnint4;
      __builtin_amdgcn_global_load_async_to_lds_b128(
          (gnint4*)(uintptr_t)src,
          (lnint4*)(uint32_t)(uintptr_t)&Abuf[ldRow][ldK], 0, 0);
      __builtin_amdgcn_global_load_async_to_lds_b128(
          (gnint4*)(uintptr_t)(src + 8),
          (lnint4*)(uint32_t)(uintptr_t)&Abuf[ldRow][ldK + 8], 0, 0);
#if __has_builtin(__builtin_amdgcn_s_wait_asynccnt)
      __builtin_amdgcn_s_wait_asynccnt(0);
#else
      asm volatile("s_wait_asynccnt 0x0" ::: "memory");
#endif
#else
#pragma unroll
      for (int i = 0; i < 16; ++i) Abuf[ldRow][ldK + i] = src[i];
#endif
    } else {
      const float* src =
          (const float*)Ap + (size_t)(rowBase + ldRow) * Kdim + k0 + ldK;
#pragma unroll
      for (int i = 0; i < 16; ++i) Abuf[ldRow][ldK + i] = (__bf16)src[i];
    }
    // Stage W^T tile: Wt[n][k] = W[k0+k][colBase+n]
#pragma unroll
    for (int i = 0; i < 8; ++i)
      Wt[wn][wk + i] = (__bf16)W[(size_t)(k0 + wk + i) * Ndim + colBase + wn];

    if (k0 + 32 < Kdim)  // hint: global_prefetch of the next W k-slab
      __builtin_prefetch(&W[(size_t)(k0 + 32 + wk) * Ndim + colBase + wn], 0, 1);

    __syncthreads();

    v16bf a = load_a_frag(&Abuf[w * 16][0], 32, 0);
#pragma unroll
    for (int t = 0; t < 4; ++t) {
      v16bf b = load_b_frag(&Wt[0][0], 32, t * 16, 0);
      acc[t] = wmma_bf16(a, b, acc[t]);
    }
    __syncthreads();
  }

  // Epilogue: C element (m,n): VGPR r -> row r+8h, lane -> col n
  const int n_ = lane & 15, h_ = lane >> 4;
#pragma unroll
  for (int t = 0; t < 4; ++t) {
    const int col = colBase + t * 16 + n_;
    const float bval = bias[col];
#pragma unroll
    for (int r = 0; r < 8; ++r) {
      const int row = rowBase + w * 16 + r + 8 * h_;
      const float val = acc[t][r] + bval;
      if (mode == 2) {
        // write-once streaming output: keep it out of L2
        __builtin_nontemporal_store(val, &((float*)Out)[(size_t)row * Ndim + col]);
      } else {
        const int bb = row >> 11, s = row & 2047;   // row = bb*SEQ + s
        const int hd = col >> 6,  d = col & 63;     // col = hd*DEPTH + d
        const size_t idx =
            (mode == 0)
                ? ((((size_t)bb * NHEAD + hd) * SEQ + s) * DEPTH + d)
                : ((((size_t)bb * NHEAD + hd) * DEPTH + d) * SEQ + s);
        ((__bf16*)Out)[idx] = (__bf16)val;
      }
    }
  }
}

// ---------------------------------------------------------------------------
// Fused attention: per block = one (b,h) and 16 query rows.
//   scores = (q @ k^T) * scale + mask * (-1e9); weights = softmax(scores)
//   weights -> d_out (fp32, NT); out = weights @ v -> attnOut (bf16, merged)
// ---------------------------------------------------------------------------
__global__ __launch_bounds__(256) void attention_kernel(
    const __bf16* __restrict__ qh, const __bf16* __restrict__ kh,
    const __bf16* __restrict__ vth, const float* __restrict__ mask,
    float* __restrict__ weightsOut, __bf16* __restrict__ attnOut) {
  __shared__ alignas(16) __bf16 stage[8][16][32];  // per-wave bf16 weight tiles
  __shared__ float wmaxS[8][16];
  __shared__ float wsumS[8][16];
  __shared__ float rowM[16];
  __shared__ float rowInv[16];
  __shared__ float outPart[8][16][64];             // per-wave PV partials

  const int tid = threadIdx.x;
  const int w = tid >> 5, lane = tid & 31;
  const int n_ = lane & 15, h_ = lane >> 4;
  const int bh = blockIdx.x;            // 0..B*H-1
  const int b = bh >> 4;
  const int head = bh & 15;
  const int q0 = blockIdx.y * 16;

  const __bf16* qb = qh + (size_t)bh * SEQ * DEPTH + (size_t)q0 * DEPTH;
  const __bf16* kb = kh + (size_t)bh * SEQ * DEPTH;
  const __bf16* vb = vth + (size_t)bh * DEPTH * SEQ;
  const float* mb = mask + (size_t)b * SEQ * SEQ + (size_t)q0 * SEQ;
  float* wo = weightsOut + (size_t)bh * SEQ * SEQ + (size_t)q0 * SEQ;

  const float scale = 0.125f;  // 1/sqrt(DEPTH)

  // q fragments live in registers for the whole kernel
  const v16bf aq0 = load_a_frag(qb, DEPTH, 0);
  const v16bf aq1 = load_a_frag(qb, DEPTH, 32);

  // -------- pass 1: online per-row (max, sum of exp) --------
  float pm[8], ps[8];
#pragma unroll
  for (int r = 0; r < 8; ++r) { pm[r] = -INFINITY; ps[r] = 0.f; }

  for (int i = 0; i < 16; ++i) {
    const int n0 = (w + 8 * i) * 16;
    v8f c = {};
    c = wmma_bf16(aq0, load_b_frag(kb, DEPTH, n0, 0), c);
    c = wmma_bf16(aq1, load_b_frag(kb, DEPTH, n0, 32), c);
#pragma unroll
    for (int r = 0; r < 8; ++r) {
      const int m = r + 8 * h_;
      const float x = c[r] * scale + mb[(size_t)m * SEQ + n0 + n_] * (-1e9f);
      const float nm = fmaxf(pm[r], x);
      ps[r] = ps[r] * __expf(pm[r] - nm) + __expf(x - nm);
      pm[r] = nm;
    }
  }
  // reduce across the 16 lanes of each half (masks 1..8 stay within a half)
#pragma unroll
  for (int off = 1; off < 16; off <<= 1) {
#pragma unroll
    for (int r = 0; r < 8; ++r) {
      const float om = __shfl_xor(pm[r], off);
      const float os = __shfl_xor(ps[r], off);
      const float nm = fmaxf(pm[r], om);
      ps[r] = ps[r] * __expf(pm[r] - nm) + os * __expf(om - nm);
      pm[r] = nm;
    }
  }
  if (n_ == 0) {
#pragma unroll
    for (int r = 0; r < 8; ++r) {
      wmaxS[w][r + 8 * h_] = pm[r];
      wsumS[w][r + 8 * h_] = ps[r];
    }
  }
  __syncthreads();
  if (tid < 16) {  // deterministic serial combine per row
    float M = -INFINITY;
#pragma unroll
    for (int ww = 0; ww < 8; ++ww) M = fmaxf(M, wmaxS[ww][tid]);
    float s = 0.f;
#pragma unroll
    for (int ww = 0; ww < 8; ++ww) s += wsumS[ww][tid] * __expf(wmaxS[ww][tid] - M);
    rowM[tid] = M;
    rowInv[tid] = 1.f / s;
  }
  __syncthreads();

  float rM[8], rI[8];
#pragma unroll
  for (int r = 0; r < 8; ++r) {
    rM[r] = rowM[r + 8 * h_];
    rI[r] = rowInv[r + 8 * h_];
  }

  // -------- pass 2: recompute scores, emit weights, accumulate weights@v ----
  v8f acc[4] = {};
  for (int i = 0; i < 8; ++i) {
    const int n0 = (w + 8 * i) * 32;  // K-chunk of 32 keys
#pragma unroll
    for (int sub = 0; sub < 2; ++sub) {
      const int nn0 = n0 + sub * 16;
      v8f c = {};
      c = wmma_bf16(aq0, load_b_frag(kb, DEPTH, nn0, 0), c);
      c = wmma_bf16(aq1, load_b_frag(kb, DEPTH, nn0, 32), c);
#pragma unroll
      for (int r = 0; r < 8; ++r) {
        const int m = r + 8 * h_;
        const float x = c[r] * scale + mb[(size_t)m * SEQ + nn0 + n_] * (-1e9f);
        const float wgt = __expf(x - rM[r]) * rI[r];
        // 536 MB streaming write (3x L2 size): non-temporal, bypass caches
        __builtin_nontemporal_store(wgt, &wo[(size_t)m * SEQ + nn0 + n_]);
        stage[w][m][sub * 16 + n_] = (__bf16)wgt;            // bf16 A staging
      }
    }
    const v16bf aw = load_a_frag(&stage[w][0][0], 32, 0);
#pragma unroll
    for (int t = 0; t < 4; ++t) {
      const v16bf bvv = load_b_frag(vb, SEQ, t * 16, n0);    // vT rows = depth
      acc[t] = wmma_bf16(aw, bvv, acc[t]);
    }
  }

  // per-wave partials, then fixed-order (deterministic) reduction
#pragma unroll
  for (int t = 0; t < 4; ++t)
#pragma unroll
    for (int r = 0; r < 8; ++r)
      outPart[w][r + 8 * h_][t * 16 + n_] = acc[t][r];
  __syncthreads();

  for (int e = tid; e < 16 * 64; e += 256) {
    const int m = e >> 6, d = e & 63;
    float s = 0.f;
#pragma unroll
    for (int ww = 0; ww < 8; ++ww) s += outPart[ww][m][d];
    attnOut[((size_t)(b * SEQ + q0 + m)) * DMODEL + head * DEPTH + d] = (__bf16)s;
  }
}

// ---------------------------------------------------------------------------
extern "C" void kernel_launch(void* const* d_in, const int* in_sizes, int n_in,
                              void* d_out, int out_size, void* d_ws,
                              size_t ws_size, hipStream_t stream) {
  const float* Q    = (const float*)d_in[0];
  const float* K    = (const float*)d_in[1];
  const float* V    = (const float*)d_in[2];
  const float* mask = (const float*)d_in[3];
  const float* Wq   = (const float*)d_in[4];
  const float* bq   = (const float*)d_in[5];
  const float* Wk   = (const float*)d_in[6];
  const float* bk   = (const float*)d_in[7];
  const float* Wv   = (const float*)d_in[8];
  const float* bv   = (const float*)d_in[9];
  const float* Wo   = (const float*)d_in[10];
  const float* bo   = (const float*)d_in[11];

  const size_t PROJ = (size_t)BATCH * SEQ * DMODEL;  // 4,194,304 elements
  __bf16* qh      = (__bf16*)d_ws;                   // [B,H,S,D] bf16
  __bf16* kh      = qh + PROJ;                       // [B,H,S,D] bf16
  __bf16* vth     = kh + PROJ;                       // [B,H,D,S] bf16
  __bf16* attnOut = vth + PROJ;                      // [B,S,DM]  bf16

  float* outF = (float*)d_out;          // final output, PROJ fp32
  float* weightsOut = outF + PROJ;      // softmax weights, B*H*S*S fp32

  dim3 gGemm(BATCH * SEQ / 128, DMODEL / 64);
  gemm_bias_kernel<false><<<gGemm, 256, 0, stream>>>(Q, Wq, bq, qh, 0);
  gemm_bias_kernel<false><<<gGemm, 256, 0, stream>>>(K, Wk, bk, kh, 0);
  gemm_bias_kernel<false><<<gGemm, 256, 0, stream>>>(V, Wv, bv, vth, 1);

  dim3 gAttn(BATCH * NHEAD, SEQ / 16);
  attention_kernel<<<gAttn, 256, 0, stream>>>(qh, kh, vth, mask, weightsOut,
                                              attnOut);

  gemm_bias_kernel<true><<<gGemm, 256, 0, stream>>>(attnOut, Wo, bo, outF, 2);
}
